// UTDE_82334523064906
// MI455X (gfx1250) — matrix operand order repaired and checked by reading, not compiled, not run on hardware
//
#include <hip/hip_runtime.h>
#include <hip/hip_bf16.h>
#include <math.h>

// mTAND attention for MI455X (gfx1250, wave32, WMMA).
// Factorization: p_attn[b,h,q,k,d] = M[b,k,d]*exp(s[b,h,q,k]-rowmax)/Z[b,h,q,d]
// (masked -10000 logits underflow to exactly 0 in fp32), so the 5-D
// softmax/einsum becomes two dense [48x128]x[128x128] f16 WMMA GEMMs per (b,h),
// plus a [384x1024]x[1024x64] WMMA output GEMM.

typedef __attribute__((ext_vector_type(16))) _Float16 v16h;
typedef __attribute__((ext_vector_type(8)))  float    v8f;

#define B_  8
#define TQ_ 48
#define TK_ 128
#define F_  64
#define ET_ 64
#define H_  8
#define DK_ 8
#define D_  128          // 2*F
#define HD_ 1024         // H*D

// ---------------- prep: masked values / mask, transposed to [D, TK] f16 -----
__global__ void prep_vals(const float* __restrict__ X, const float* __restrict__ mask,
                          _Float16* __restrict__ XmT, _Float16* __restrict__ MT) {
    int idx = blockIdx.x * blockDim.x + threadIdx.x;   // B*D*TK = 131072
    if (idx >= B_ * D_ * TK_) return;
    int b = idx / (D_ * TK_);
    int r = idx % (D_ * TK_);
    int d = r / TK_;
    int k = r % TK_;
    int f = (d < F_) ? d : d - F_;
    float mv = mask[(b * TK_ + k) * F_ + f];
    float xv = (d < F_) ? X[(b * TK_ + k) * F_ + d] * mv : mv;
    XmT[idx] = (_Float16)xv;
    MT[idx]  = (_Float16)mv;
}

// ---------------- prep: Wo transposed to [ET, HD] f16 -----------------------
__global__ void prep_wot(const float* __restrict__ Wo, _Float16* __restrict__ WoT) {
    int idx = blockIdx.x * blockDim.x + threadIdx.x;   // ET*HD = 65536
    if (idx >= ET_ * HD_) return;
    int n = idx / HD_;
    int k = idx % HD_;
    WoT[idx] = (_Float16)Wo[k * ET_ + n];
}

// ---------------- prep: mean of X_irg over k (uniform-softmax fallback) -----
__global__ void prep_xmean(const float* __restrict__ X, const float* __restrict__ mask,
                           float* __restrict__ Xmean) {
    int idx = blockIdx.x * blockDim.x + threadIdx.x;   // B*D = 1024
    if (idx >= B_ * D_) return;
    int b = idx / D_;
    int d = idx % D_;
    float s = 0.f;
    for (int k = 0; k < TK_; ++k)
        s += (d < F_) ? X[(b * TK_ + k) * F_ + d] : mask[(b * TK_ + k) * F_ + (d - F_)];
    Xmean[idx] = s * (1.0f / TK_);
}

// ---------------- time embed + linear projection (q or k path) --------------
// one 64-thread block per row; emb built cooperatively in LDS, then 64x64 matvec
__global__ void embed_proj(const float* __restrict__ t_in,
                           const float* __restrict__ w_lin, const float* __restrict__ b_lin,
                           const float* __restrict__ w_per, const float* __restrict__ b_per,
                           const float* __restrict__ W, const float* __restrict__ bias,
                           float* __restrict__ out) {
    __shared__ float emb[ET_];
    int row = blockIdx.x;
    int j = threadIdx.x;
    float t = t_in[row];
    if (j == 0) emb[0] = t * w_lin[0] + b_lin[0];
    else        emb[j] = sinf(t * w_per[j - 1] + b_per[j - 1]);
    __syncthreads();
    float acc = bias[j];
    #pragma unroll 8
    for (int i = 0; i < ET_; ++i) acc += emb[i] * W[i * ET_ + j];
    out[row * ET_ + j] = acc;
}

// ---------------- scores + exp(s - rowmax) -> E f16 [B*H, TQ, TK] -----------
__global__ void scores_exp(const float* __restrict__ qp, const float* __restrict__ kp,
                           _Float16* __restrict__ E) {
    __shared__ float red[TK_];
    __shared__ float qv[DK_];
    int bid = blockIdx.x;          // (b*H + h)*TQ + q
    int q   = bid % TQ_;
    int bh  = bid / TQ_;
    int h   = bh % H_;
    int b   = bh / H_;
    int k   = threadIdx.x;         // 128 threads
    if (k < DK_) qv[k] = qp[(b * TQ_ + q) * ET_ + h * DK_ + k];
    __syncthreads();
    const float* kr = kp + (b * TK_ + k) * ET_ + h * DK_;
    float s = 0.f;
    #pragma unroll
    for (int d = 0; d < DK_; ++d) s += qv[d] * kr[d];
    s *= 0.35355339059327373f;     // 1/sqrt(DK)
    red[k] = s;
    __syncthreads();
    for (int off = 64; off > 0; off >>= 1) {
        if (k < off) red[k] = fmaxf(red[k], red[k + off]);
        __syncthreads();
    }
    float e = __expf(s - red[0]);
    E[((size_t)bh * TQ_ + q) * TK_ + k] = (_Float16)e;
}

// ---------------- attention GEMMs: N = E@Xm, Z = E@M, out = N/Z -------------
// one wave per 16x16 output tile pair; 1536 waves total (64 bh * 3 mt * 8 nt)
__global__ void attn_wmma(const _Float16* __restrict__ E,
                          const _Float16* __restrict__ XmT,
                          const _Float16* __restrict__ MT,
                          const float* __restrict__ Xmean,
                          _Float16* __restrict__ Xcat) {
    int wave = (blockIdx.x * blockDim.x + threadIdx.x) >> 5;
    int lane = threadIdx.x & 31;
    int bh = wave / 24;            // b*H + h
    int t  = wave % 24;
    int mt = t >> 3;               // 0..2 (TQ tiles)
    int nt = t & 7;                // 0..7 (D tiles)
    int b  = bh >> 3;
    int h  = bh & 7;
    int lhalf = lane & 15;
    int lhi   = lane >> 4;         // 0 or 1

    const _Float16* Erow = E   + ((size_t)bh * TQ_ + mt * 16 + lhalf) * TK_;
    const _Float16* Xrow = XmT + ((size_t)b  * D_  + nt * 16 + lhalf) * TK_;
    const _Float16* Mrow = MT  + ((size_t)b  * D_  + nt * 16 + lhalf) * TK_;

    v8f accN = {};
    v8f accZ = {};
    #pragma unroll
    for (int kk = 0; kk < TK_; kk += 32) {
        union { v16h v; uint4 u[2]; } a, bx, bm;
        // A frag: lane holds row (lane&15), K {0..15} (+16 hi-half), contiguous
        a.u[0] = *(const uint4*)(Erow + kk + lhi * 16);
        a.u[1] = *(const uint4*)(Erow + kk + lhi * 16 + 8);
        // B frag: lane holds col (lane&15), K {0..15} (+16 hi-half), contiguous
        const uint4* bp = (const uint4*)(Xrow + kk + lhi * 16);
        bx.u[0] = bp[0]; bx.u[1] = bp[1];
        const uint4* mp = (const uint4*)(Mrow + kk + lhi * 16);
        bm.u[0] = mp[0]; bm.u[1] = mp[1];
        accN = __builtin_amdgcn_wmma_f32_16x16x32_f16(false, a.v, false, bx.v,
                                                      (short)0, accN, false, false);
        accZ = __builtin_amdgcn_wmma_f32_16x16x32_f16(false, a.v, false, bm.v,
                                                      (short)0, accZ, false, false);
    }
    // C frag: VGPR v -> row v + 8*lhi, col lane&15. Write f16 into Xcat[B*TQ, H*D]
    int d  = nt * 16 + lhalf;
    float xm = Xmean[b * D_ + d];
    #pragma unroll
    for (int v = 0; v < 8; ++v) {
        int q = mt * 16 + v + lhi * 8;
        float z = accZ[v];
        float o = (z > 0.f) ? accN[v] / z : xm;   // all-masked column -> uniform mean
        Xcat[((size_t)(b * TQ_ + q)) * HD_ + h * D_ + d] = (_Float16)o;
    }
}

// ---------------- output GEMM: [384,1024] @ [1024,64] + bo ------------------
// 96 waves (24 mtiles x 4 ntiles)
__global__ void out_wmma(const _Float16* __restrict__ Xcat,
                         const _Float16* __restrict__ WoT,
                         const float* __restrict__ bo,
                         float* __restrict__ out) {
    int wave = (blockIdx.x * blockDim.x + threadIdx.x) >> 5;
    int lane = threadIdx.x & 31;
    int mt = wave >> 2;            // 0..23
    int nt = wave & 3;             // 0..3
    int lhalf = lane & 15;
    int lhi   = lane >> 4;

    const _Float16* Arow = Xcat + ((size_t)(mt * 16 + lhalf)) * HD_;
    const _Float16* Brow = WoT  + ((size_t)(nt * 16 + lhalf)) * HD_;

    v8f acc = {};
    for (int kk = 0; kk < HD_; kk += 32) {
        union { v16h v; uint4 u[2]; } a, bb;
        a.u[0] = *(const uint4*)(Arow + kk + lhi * 16);
        a.u[1] = *(const uint4*)(Arow + kk + lhi * 16 + 8);
        const uint4* bp = (const uint4*)(Brow + kk + lhi * 16);
        bb.u[0] = bp[0]; bb.u[1] = bp[1];
        acc = __builtin_amdgcn_wmma_f32_16x16x32_f16(false, a.v, false, bb.v,
                                                     (short)0, acc, false, false);
    }
    int n = nt * 16 + lhalf;
    float bias = bo[n];
    #pragma unroll
    for (int v = 0; v < 8; ++v) {
        int row = mt * 16 + v + lhi * 8;
        out[row * ET_ + n] = acc[v] + bias;
    }
}

// ---------------------------------------------------------------------------
extern "C" void kernel_launch(void* const* d_in, const int* in_sizes, int n_in,
                              void* d_out, int out_size, void* d_ws, size_t ws_size,
                              hipStream_t stream) {
    const float* X      = (const float*)d_in[0];
    const float* mask_X = (const float*)d_in[1];
    const float* T      = (const float*)d_in[2];
    const float* alpha  = (const float*)d_in[3];
    const float* w_lin  = (const float*)d_in[4];
    const float* b_lin  = (const float*)d_in[5];
    const float* w_per  = (const float*)d_in[6];
    const float* b_per  = (const float*)d_in[7];
    const float* Wq     = (const float*)d_in[8];
    const float* bq     = (const float*)d_in[9];
    const float* Wk     = (const float*)d_in[10];
    const float* bk     = (const float*)d_in[11];
    const float* Wo     = (const float*)d_in[12];
    const float* bo     = (const float*)d_in[13];
    float* out = (float*)d_out;

    // workspace layout (all offsets 256-B aligned)
    char* ws = (char*)d_ws;
    float*    q_proj = (float*)(ws + 0);             //  384*64 f32   =  98304 B
    float*    k_proj = (float*)(ws + 98304);         // 1024*64 f32   = 262144 B
    _Float16* E      = (_Float16*)(ws + 360448);     // 64*48*128 f16 = 786432 B
    _Float16* XmT    = (_Float16*)(ws + 1146880);    // 8*128*128 f16 = 262144 B
    _Float16* MT     = (_Float16*)(ws + 1409024);    // 8*128*128 f16 = 262144 B
    _Float16* WoT    = (_Float16*)(ws + 1671168);    // 64*1024  f16  = 131072 B
    _Float16* Xcat   = (_Float16*)(ws + 1802240);    // 384*1024 f16  = 786432 B
    float*    Xmean  = (float*)(ws + 2588672);       // 1024 f32      =   4096 B
    (void)ws_size; (void)in_sizes; (void)n_in; (void)out_size;

    prep_vals <<<(B_ * D_ * TK_ + 255) / 256, 256, 0, stream>>>(X, mask_X, XmT, MT);
    prep_wot  <<<(ET_ * HD_ + 255) / 256,     256, 0, stream>>>(Wo, WoT);
    prep_xmean<<<(B_ * D_ + 255) / 256,       256, 0, stream>>>(X, mask_X, Xmean);

    embed_proj<<<B_ * TQ_, ET_, 0, stream>>>(alpha, w_lin, b_lin, w_per, b_per,
                                             Wq, bq, q_proj);
    embed_proj<<<B_ * TK_, ET_, 0, stream>>>(T, w_lin, b_lin, w_per, b_per,
                                             Wk, bk, k_proj);

    scores_exp<<<B_ * H_ * TQ_, TK_, 0, stream>>>(q_proj, k_proj, E);

    // 64 bh * 24 tiles = 1536 waves; 8 waves/block -> 192 blocks
    attn_wmma<<<192, 256, 0, stream>>>(E, XmT, MT, Xmean, Xcat);

    // 96 waves -> 12 blocks
    out_wmma<<<12, 256, 0, stream>>>(Xcat, WoT, bo, out);
}
